// Siglip2Attention_23733989278428
// MI455X (gfx1250) — matrix operand order repaired and checked by reading, not compiled, not run on hardware
//
#include <hip/hip_runtime.h>
#include <hip/hip_bf16.h>

// ---------------------------------------------------------------------------
// Types for WMMA fragments (gfx1250, wave32)
// ---------------------------------------------------------------------------
typedef __attribute__((ext_vector_type(4)))  unsigned short v4us;
typedef __attribute__((ext_vector_type(8)))  unsigned short v8us;
typedef __attribute__((ext_vector_type(16))) unsigned short v16us;
typedef __attribute__((ext_vector_type(16))) __bf16        v16bf;
typedef __attribute__((ext_vector_type(8)))  float         v8f;

#define SEQ 4096
#define EMB 1152
#define NH  16
#define HD  72
#define NSEG 8

// float -> bf16 bits, round-to-nearest-even
__device__ __forceinline__ unsigned short f2bf(float f) {
    unsigned u = __builtin_bit_cast(unsigned, f);
    u += 0x7FFFu + ((u >> 16) & 1u);
    return (unsigned short)(u >> 16);
}

// Build a 16-element bf16 fragment from two 16-byte LDS chunks
__device__ __forceinline__ v16bf ld_frag(const unsigned short* lo,
                                         const unsigned short* hi) {
    v8us a = *(const v8us*)lo;
    v8us b = *(const v8us*)hi;
    v16us t;
#pragma unroll
    for (int i = 0; i < 8; ++i) { t[i] = a[i]; t[i + 8] = b[i]; }
    return __builtin_bit_cast(v16bf, t);
}

__device__ __forceinline__ v8f wmma_bf16(v16bf a, v16bf b, v8f c) {
    return __builtin_amdgcn_wmma_f32_16x16x32_bf16(
        false, a, false, b, (short)0, c, false, false);
}

// Async 16-byte global -> LDS copy (per-lane), tracked by ASYNCcnt.
__device__ __forceinline__ void async_ld16(const void* lds_ptr, const void* gptr) {
    unsigned lds = (unsigned)(unsigned long long)lds_ptr;   // low 32 bits = LDS offset
    asm volatile("global_load_async_to_lds_b128 %0, %1, off"
                 :: "v"(lds), "v"(gptr) : "memory");
}
#define WAIT_ASYNC(N) asm volatile("s_wait_asynccnt " #N ::: "memory")

// ---------------------------------------------------------------------------
// f32 -> bf16 bulk cast (float4 / v4us vectorized), n4 = elements/4
// ---------------------------------------------------------------------------
__global__ __launch_bounds__(256) void cast_bf16(
    const float* __restrict__ src, unsigned short* __restrict__ dst, int n4)
{
    int i = blockIdx.x * 256 + threadIdx.x;
    if (i >= n4) return;
    float4 f = ((const float4*)src)[i];
    v4us p;
    p[0] = f2bf(f.x); p[1] = f2bf(f.y); p[2] = f2bf(f.z); p[3] = f2bf(f.w);
    ((v4us*)dst)[i] = p;
}

// ---------------------------------------------------------------------------
// GEMM: out[s, n] = sum_k Xb[s,k] * Wb[n,k] + bias[n]   (bf16 in, f32 out)
// 128(M) x 64(N) tile, 256 threads = 8 waves; double-buffered async LDS
// staging: each thread issues 3x16B async copies per k-step.
// grid = (E/64, S/128)
// ---------------------------------------------------------------------------
__global__ __launch_bounds__(256) void gemm_bf16(
    const unsigned short* __restrict__ Xb, const unsigned short* __restrict__ Wb,
    const float* __restrict__ bias, float* __restrict__ out)
{
    __shared__ __align__(16) unsigned short sA[2][128][40];  // [m][k]
    __shared__ __align__(16) unsigned short sB[2][64][40];   // [n][k]

    const int tid = threadIdx.x;
    const int m0  = blockIdx.y * 128;
    const int n0  = blockIdx.x * 64;
    const int wv  = tid >> 5;
    const int ln  = tid & 15;
    const int hf  = (tid >> 4) & 1;

    const int ca0 = tid, ca1 = tid + 256;   // A-tile 16B chunk ids (512 total)
    const int cb0 = tid;                    // B-tile 16B chunk ids (256 total)

    v8f acc[4];
#pragma unroll
    for (int nt = 0; nt < 4; ++nt)
#pragma unroll
        for (int j = 0; j < 8; ++j) acc[nt][j] = 0.f;

    // prologue: stage tile 0 into buffer 0
    async_ld16(&sA[0][ca0 >> 2][(ca0 & 3) * 8],
               Xb + (size_t)(m0 + (ca0 >> 2)) * EMB + (ca0 & 3) * 8);
    async_ld16(&sA[0][ca1 >> 2][(ca1 & 3) * 8],
               Xb + (size_t)(m0 + (ca1 >> 2)) * EMB + (ca1 & 3) * 8);
    async_ld16(&sB[0][cb0 >> 2][(cb0 & 3) * 8],
               Wb + (size_t)(n0 + (cb0 >> 2)) * EMB + (cb0 & 3) * 8);

    const int NT = EMB / 32;   // 36 k-steps
    for (int it = 0; it < NT; ++it) {
        const int cur = it & 1;
        if (it + 1 < NT) {
            const int k1 = (it + 1) * 32;
            async_ld16(&sA[1 - cur][ca0 >> 2][(ca0 & 3) * 8],
                       Xb + (size_t)(m0 + (ca0 >> 2)) * EMB + k1 + (ca0 & 3) * 8);
            async_ld16(&sA[1 - cur][ca1 >> 2][(ca1 & 3) * 8],
                       Xb + (size_t)(m0 + (ca1 >> 2)) * EMB + k1 + (ca1 & 3) * 8);
            async_ld16(&sB[1 - cur][cb0 >> 2][(cb0 & 3) * 8],
                       Wb + (size_t)(n0 + (cb0 >> 2)) * EMB + k1 + (cb0 & 3) * 8);
            if (it + 2 < NT)   // deep prefetch into GL2
                __builtin_prefetch(Xb + (size_t)(m0 + (ca0 >> 2)) * EMB + (it + 2) * 32, 0, 1);
            WAIT_ASYNC(3);     // tile `cur` complete; next tile still in flight
        } else {
            WAIT_ASYNC(0);
        }
        __syncthreads();

        v16bf afrag = ld_frag(&sA[cur][wv * 16 + ln][16 * hf],
                              &sA[cur][wv * 16 + ln][16 * hf + 8]);
#pragma unroll
        for (int nt = 0; nt < 4; ++nt) {
            v16bf bfrag = ld_frag(&sB[cur][nt * 16 + ln][16 * hf],
                                  &sB[cur][nt * 16 + ln][16 * hf + 8]);
            acc[nt] = wmma_bf16(afrag, bfrag, acc[nt]);
        }
        __syncthreads();       // done reading buf `cur` before it is re-filled
    }

#pragma unroll
    for (int nt = 0; nt < 4; ++nt) {
        int col  = n0 + nt * 16 + ln;
        float bv = bias[col];
#pragma unroll
        for (int j = 0; j < 8; ++j) {
            int row = m0 + wv * 16 + j + 8 * hf;
            out[(size_t)row * EMB + col] = acc[nt][j] + bv;
        }
    }
}

// ---------------------------------------------------------------------------
// RoPE on f32 q/k, emitting bf16 copies for the attention kernel
// ---------------------------------------------------------------------------
__global__ __launch_bounds__(256) void rope_bf16(
    const float* __restrict__ q, const float* __restrict__ k,
    const float* __restrict__ cp, const float* __restrict__ sp,
    unsigned short* __restrict__ qb, unsigned short* __restrict__ kb)
{
    int idx = blockIdx.x * 256 + threadIdx.x;
    const int total = SEQ * NH * (HD / 2);
    if (idx >= total) return;
    int p  = idx % (HD / 2);
    int t  = idx / (HD / 2);
    int hh = t % NH;
    int s  = t / NH;
    size_t base = (size_t)s * EMB + hh * HD;
    float c1 = cp[(size_t)s * HD + p];
    float s1 = sp[(size_t)s * HD + p];
    float c2 = cp[(size_t)s * HD + p + HD / 2];
    float s2 = sp[(size_t)s * HD + p + HD / 2];

    float qx1 = q[base + p], qx2 = q[base + p + HD / 2];
    qb[base + p]          = f2bf(qx1 * c1 - qx2 * s1);
    qb[base + p + HD / 2] = f2bf(qx2 * c2 + qx1 * s2);

    float kx1 = k[base + p], kx2 = k[base + p + HD / 2];
    kb[base + p]          = f2bf(kx1 * c1 - kx2 * s1);
    kb[base + p + HD / 2] = f2bf(kx2 * c2 + kx1 * s2);
}

// ---------------------------------------------------------------------------
// V (S,E) f32  ->  Vt (E,S) bf16  (32x32 LDS tile transpose)
// grid = (E/32, S/32)
// ---------------------------------------------------------------------------
__global__ __launch_bounds__(256) void vt_kernel(
    const float* __restrict__ V, unsigned short* __restrict__ vt)
{
    __shared__ float sT[32][33];
    const int e0 = blockIdx.x * 32;
    const int s0 = blockIdx.y * 32;
    const int r  = threadIdx.x >> 5;
    const int c  = threadIdx.x & 31;
#pragma unroll
    for (int i = 0; i < 4; ++i)
        sT[r + 8 * i][c] = V[(size_t)(s0 + r + 8 * i) * EMB + e0 + c];
    __syncthreads();
#pragma unroll
    for (int i = 0; i < 4; ++i)
        vt[(size_t)(e0 + r + 8 * i) * SEQ + s0 + c] = f2bf(sT[c][r + 8 * i]);
}

// ---------------------------------------------------------------------------
// Block-diagonal flash attention, all operands bf16 + async LDS staging.
// 1 WG = (64 query rows, 1 head); 128 threads = 4 waves. D=72 padded to 96.
// Writes bf16 output (feeds O-projection GEMM). grid = (S/64, H)
// ---------------------------------------------------------------------------
__global__ __launch_bounds__(128) void attn_kernel(
    const unsigned short* __restrict__ Qb, const unsigned short* __restrict__ Kb,
    const unsigned short* __restrict__ Vt, const int* __restrict__ cu,
    unsigned short* __restrict__ Ob)
{
    __shared__ __align__(16) unsigned short sQ[64][104];   // [m][d]
    __shared__ __align__(16) unsigned short sK[64][104];   // [t][d]
    __shared__ __align__(16) unsigned short sVt[80][72];   // [d][t]
    __shared__ __align__(16) unsigned short sP[64][72];    // [m][t]

    const int tid  = threadIdx.x;
    const int wv   = tid >> 5;
    const int ln   = tid & 15;
    const int hf   = (tid >> 4) & 1;
    const int head = blockIdx.y;
    const int r0   = blockIdx.x * 64;
    const float scale = 0.117851130f;   // 72^-0.5

    int kstart = 0, kend = SEQ;
#pragma unroll
    for (int i = 0; i < NSEG; ++i) {
        int a = cu[i], b = cu[i + 1];
        if (r0 >= a && r0 < b) { kstart = a; kend = b; }
    }

    // zero padding regions once (d in [72,96) for Q/K, d in [72,80) for Vt)
    for (int idx = tid; idx < 64 * 24; idx += 128) {
        int r = idx / 24, c = 72 + idx % 24;
        sQ[r][c] = 0; sK[r][c] = 0;
    }
    for (int idx = tid; idx < 8 * 64; idx += 128)
        sVt[72 + (idx >> 6)][idx & 63] = 0;

    // async-stage Q tile (64 rows x 144B = 9 chunks/row)
    for (int idx = tid; idx < 64 * 9; idx += 128) {
        int r = idx / 9, c = idx - r * 9;
        async_ld16(&sQ[r][c * 8],
                   Qb + (size_t)(r0 + r) * EMB + head * HD + c * 8);
    }

    float m[8], l[8];
    v8f o[5];
#pragma unroll
    for (int j = 0; j < 8; ++j) { m[j] = -3.0e38f; l[j] = 0.f; }
#pragma unroll
    for (int nt = 0; nt < 5; ++nt)
#pragma unroll
        for (int j = 0; j < 8; ++j) o[nt][j] = 0.f;

    for (int t0 = kstart; t0 < kend; t0 += 64) {
        int tv = kend - t0; if (tv > 64) tv = 64;
        __syncthreads();   // previous chunk fully consumed before restaging

        for (int idx = tid; idx < 64 * 9; idx += 128) {        // K chunk [t][d]
            int r = idx / 9, c = idx - r * 9;
            int row = t0 + r; if (row > kend - 1) row = kend - 1;
            async_ld16(&sK[r][c * 8],
                       Kb + (size_t)row * EMB + head * HD + c * 8);
        }
        for (int idx = tid; idx < 72 * 8; idx += 128) {        // V chunk [d][t]
            int d = idx >> 3, c = idx & 7;
            async_ld16(&sVt[d][c * 8],
                       Vt + (size_t)(head * HD + d) * SEQ + t0 + c * 8);
        }
        WAIT_ASYNC(0);
        __syncthreads();

        // ---- scores = Q . K^T ----
        v8f sc[4];
#pragma unroll
        for (int nt = 0; nt < 4; ++nt)
#pragma unroll
            for (int j = 0; j < 8; ++j) sc[nt][j] = 0.f;

#pragma unroll
        for (int ks = 0; ks < 3; ++ks) {
            v16bf aq = ld_frag(&sQ[wv * 16 + ln][32 * ks + 16 * hf],
                               &sQ[wv * 16 + ln][32 * ks + 16 * hf + 8]);
#pragma unroll
            for (int nt = 0; nt < 4; ++nt) {
                v16bf bk = ld_frag(&sK[nt * 16 + ln][32 * ks + 16 * hf],
                                   &sK[nt * 16 + ln][32 * ks + 16 * hf + 8]);
                sc[nt] = wmma_bf16(aq, bk, sc[nt]);
            }
        }

        // ---- online softmax ----
#pragma unroll
        for (int j = 0; j < 8; ++j) {
            float mx = -3.0e38f;
#pragma unroll
            for (int nt = 0; nt < 4; ++nt) {
                int t  = nt * 16 + ln;
                float s = (t < tv) ? sc[nt][j] * scale : -3.0e38f;
                sc[nt][j] = s;
                mx = fmaxf(mx, s);
            }
#pragma unroll
            for (int d = 1; d < 16; d <<= 1)
                mx = fmaxf(mx, __shfl_xor(mx, d, 32));
            float mn = fmaxf(m[j], mx);
            float sum = 0.f;
#pragma unroll
            for (int nt = 0; nt < 4; ++nt) {
                float p = __expf(sc[nt][j] - mn);
                sc[nt][j] = p;
                sum += p;
            }
#pragma unroll
            for (int d = 1; d < 16; d <<= 1)
                sum += __shfl_xor(sum, d, 32);
            float corr = __expf(m[j] - mn);
            l[j] = l[j] * corr + sum;
            m[j] = mn;
#pragma unroll
            for (int nt = 0; nt < 5; ++nt) o[nt][j] *= corr;
        }

        // ---- P back to LDS in A-matrix layout ----
#pragma unroll
        for (int nt = 0; nt < 4; ++nt)
#pragma unroll
            for (int j = 0; j < 8; ++j)
                sP[wv * 16 + j + 8 * hf][nt * 16 + ln] = f2bf(sc[nt][j]);
        __syncthreads();

        // ---- O += P . V ----
#pragma unroll
        for (int ks = 0; ks < 2; ++ks) {
            v16bf ap = ld_frag(&sP[wv * 16 + ln][32 * ks + 16 * hf],
                               &sP[wv * 16 + ln][32 * ks + 16 * hf + 8]);
#pragma unroll
            for (int nt = 0; nt < 5; ++nt) {
                v16bf bv = ld_frag(&sVt[nt * 16 + ln][32 * ks + 16 * hf],
                                   &sVt[nt * 16 + ln][32 * ks + 16 * hf + 8]);
                o[nt] = wmma_bf16(ap, bv, o[nt]);
            }
        }
    }

    // ---- epilogue: normalize, cast bf16, store ----
#pragma unroll
    for (int nt = 0; nt < 5; ++nt) {
        int d = nt * 16 + ln;
        if (d < HD) {
#pragma unroll
            for (int j = 0; j < 8; ++j) {
                int row = r0 + wv * 16 + j + 8 * hf;
                Ob[(size_t)row * EMB + head * HD + d] = f2bf(o[nt][j] / l[j]);
            }
        }
    }
}

// ---------------------------------------------------------------------------
// Launch
// ---------------------------------------------------------------------------
extern "C" void kernel_launch(void* const* d_in, const int* in_sizes, int n_in,
                              void* d_out, int out_size, void* d_ws, size_t ws_size,
                              hipStream_t stream) {
    (void)in_sizes; (void)n_in; (void)out_size; (void)ws_size;
    const float* hs   = (const float*)d_in[0];
    const int*   cu   = (const int*)d_in[1];
    const float* cosp = (const float*)d_in[2];
    const float* sinp = (const float*)d_in[3];
    const float* Wq   = (const float*)d_in[4];
    const float* bq   = (const float*)d_in[5];
    const float* Wk   = (const float*)d_in[6];
    const float* bk   = (const float*)d_in[7];
    const float* Wv   = (const float*)d_in[8];
    const float* bv   = (const float*)d_in[9];
    const float* Wo   = (const float*)d_in[10];
    const float* bo   = (const float*)d_in[11];
    float* out = (float*)d_out;

    const size_t SE = (size_t)SEQ * EMB;
    const size_t EE = (size_t)EMB * EMB;

    char* w = (char*)d_ws;
    auto carve = [&](size_t bytes) {
        char* p = w; w += (bytes + 255) & ~(size_t)255; return p;
    };
    float*          qf    = (float*)         carve(SE * 4);
    float*          kf    = (float*)         carve(SE * 4);
    float*          vf    = (float*)         carve(SE * 4);
    unsigned short* Xb    = (unsigned short*)carve(SE * 2);
    unsigned short* Wqb   = (unsigned short*)carve(EE * 2);
    unsigned short* Wkb   = (unsigned short*)carve(EE * 2);
    unsigned short* Wvb   = (unsigned short*)carve(EE * 2);
    unsigned short* Wob   = (unsigned short*)carve(EE * 2);
    unsigned short* qb    = (unsigned short*)carve(SE * 2);
    unsigned short* kb    = (unsigned short*)carve(SE * 2);
    unsigned short* vt    = (unsigned short*)carve(SE * 2);
    unsigned short* attnb = (unsigned short*)carve(SE * 2);

    const int n4SE = (int)(SE / 4), n4EE = (int)(EE / 4);
    cast_bf16<<<(n4SE + 255) / 256, 256, 0, stream>>>(hs, Xb, n4SE);
    cast_bf16<<<(n4EE + 255) / 256, 256, 0, stream>>>(Wq, Wqb, n4EE);
    cast_bf16<<<(n4EE + 255) / 256, 256, 0, stream>>>(Wk, Wkb, n4EE);
    cast_bf16<<<(n4EE + 255) / 256, 256, 0, stream>>>(Wv, Wvb, n4EE);
    cast_bf16<<<(n4EE + 255) / 256, 256, 0, stream>>>(Wo, Wob, n4EE);

    dim3 gg(EMB / 64, SEQ / 128);           // (18, 32)
    gemm_bf16<<<gg, 256, 0, stream>>>(Xb, Wqb, bq, qf);
    gemm_bf16<<<gg, 256, 0, stream>>>(Xb, Wkb, bk, kf);
    gemm_bf16<<<gg, 256, 0, stream>>>(Xb, Wvb, bv, vf);

    const int total = SEQ * NH * (HD / 2);
    rope_bf16<<<(total + 255) / 256, 256, 0, stream>>>(qf, kf, cosp, sinp, qb, kb);
    vt_kernel<<<dim3(EMB / 32, SEQ / 32), 256, 0, stream>>>(vf, vt);

    attn_kernel<<<dim3(SEQ / 64, NH), 128, 0, stream>>>(qb, kb, vt, cu, attnb);

    gemm_bf16<<<gg, 256, 0, stream>>>(attnb, Wob, bo, out);
}
